// APM_16733192585590
// MI455X (gfx1250) — matrix operand order repaired and checked by reading, not compiled, not run on hardware
//
#include <hip/hip_runtime.h>
#include <hip/hip_bf16.h>

typedef __attribute__((ext_vector_type(16))) __bf16 v16bf;
typedef __attribute__((ext_vector_type(8)))  float  v8f;

#define FMD 256
#define ROWS 16

// ---------------- Kernel 1: G = V @ V^T (bf16), s[k] = sum_j V[k,j]^2 -------
__global__ __launch_bounds__(256) void fm_prep_kernel(const float* __restrict__ V,
                                                      __bf16* __restrict__ Gb,
                                                      float*  __restrict__ s)
{
    __shared__ float vk[FMD];
    __shared__ float red[FMD];
    const int k = blockIdx.x;
    const int n = threadIdx.x;
    vk[n] = V[k * FMD + n];
    __syncthreads();
    const float* Vn = V + (size_t)n * FMD;
    float acc = 0.f;
#pragma unroll 8
    for (int j = 0; j < FMD; ++j) acc += vk[j] * Vn[j];
    Gb[(size_t)k * FMD + n] = (__bf16)acc;
    red[n] = vk[n] * vk[n];
    __syncthreads();
    for (int off = 128; off > 0; off >>= 1) {
        if (n < off) red[n] += red[n + off];
        __syncthreads();
    }
    if (n == 0) s[k] = red[0];
}

// ---------------- Kernel 2: fused gather + WMMA GEMM + FM reduction ---------
__global__ __launch_bounds__(128) void fm_main_kernel(
    const int*   __restrict__ uid_batch,
    const int*   __restrict__ iid_batch,
    const int*   __restrict__ n_users_p,
    const float* __restrict__ user_table,
    const float* __restrict__ item_table,
    const float* __restrict__ word_table,
    const float* __restrict__ w_lin,
    const float* __restrict__ bias_u,
    const float* __restrict__ bias_i,
    const __bf16* __restrict__ Gb,
    const float* __restrict__ s,
    float*       __restrict__ out)
{
    __shared__ float  xf[ROWS][FMD];   // fp32 x tile (16 KB)
    __shared__ __bf16 xh[ROWS][FMD];   // bf16 x tile for WMMA A (8 KB)
    __shared__ float  ts[ROWS][FMD];   // t = x @ G tile (16 KB)

    const int b0  = blockIdx.x * ROWS;
    const int tid = threadIdx.x;
    const int n_users = *n_users_p;

    // ---- Phase 1: gather x = [u_e, u_g, i_e, i_g] into LDS (fp32 + bf16) ----
    {
        const int pair = tid >> 1;         // 0..63 : (row, segment)
        const int row  = pair >> 2;        // 0..15
        const int seg  = pair & 3;         // 0..3
        const int j0   = (tid & 1) * 32;   // half-segment
        const int uid  = uid_batch[b0 + row];
        const int iid  = iid_batch[b0 + row];
        const float* src;
        if (seg == 0)      src = user_table + (size_t)uid * 64;
        else if (seg == 1) src = word_table + (size_t)uid * 64;
        else if (seg == 2) src = item_table + (size_t)iid * 64;
        else               src = word_table + (size_t)(iid + n_users) * 64;
#pragma unroll
        for (int j = j0; j < j0 + 32; j += 4) {
            float4 v = *(const float4*)(src + j);
            const int kk = seg * 64 + j;
            xf[row][kk + 0] = v.x; xf[row][kk + 1] = v.y;
            xf[row][kk + 2] = v.z; xf[row][kk + 3] = v.w;
            xh[row][kk + 0] = (__bf16)v.x; xh[row][kk + 1] = (__bf16)v.y;
            xh[row][kk + 2] = (__bf16)v.z; xh[row][kk + 3] = (__bf16)v.w;
        }
    }
    __syncthreads();

    // ---- Phase 2: t[16,256] = x[16,256] @ G[256,256] via bf16 WMMA ----------
    {
        const int wave  = tid >> 5;              // 0..3 -> 4 N-tiles each
        const int lane  = tid & 31;
        const int mrow  = lane & 15;             // A: lane -> matrix row
        const int ahalf = (lane < 16) ? 0 : 8;   // A: K sub-chunk select
        const int bhalf = (lane < 16) ? 0 : 8;   // B: K sub-chunk select
        v8f acc[4] = {};
        for (int k0 = 0; k0 < FMD; k0 += 32) {
            union { uint4 u[2]; v16bf v; } A;
            A.u[0] = *(const uint4*)&xh[mrow][k0 + ahalf];
            A.u[1] = *(const uint4*)&xh[mrow][k0 + ahalf + 16];
#pragma unroll
            for (int i = 0; i < 4; ++i) {
                const int n = ((wave << 2) + i) * 16 + (lane & 15);
                union { uint4 u[2]; v16bf v; } B;
                // G is symmetric: G[k][n] == G[n][k] -> two 8-wide K chunks per lane (table order)
                const __bf16* gp = Gb + (size_t)n * FMD + k0 + bhalf;
                B.u[0] = *(const uint4*)gp;
                B.u[1] = *(const uint4*)(gp + 16);
                acc[i] = __builtin_amdgcn_wmma_f32_16x16x32_bf16(
                    false, A.v, false, B.v, (short)0, acc[i], false, false);
            }
        }
        const int mbase = (lane < 16) ? 0 : 8;
#pragma unroll
        for (int i = 0; i < 4; ++i) {
            const int n = ((wave << 2) + i) * 16 + (lane & 15);
#pragma unroll
            for (int r = 0; r < 8; ++r) ts[mbase + r][n] = acc[i][r];
        }
    }
    __syncthreads();

    // ---- Phase 3: rate = x.w + 0.5*(x.t - x^2.s) + biases + 3 ---------------
    {
        const int m  = tid >> 3;        // row 0..15
        const int c  = tid & 7;         // 8 chunks of 32 columns
        const int k0 = c * 32;
        float lin = 0.f, quad = 0.f, p2 = 0.f;
#pragma unroll 8
        for (int k = k0; k < k0 + 32; ++k) {
            const float xv = xf[m][k];
            lin  += xv * w_lin[k];
            quad += xv * ts[m][k];
            p2   += xv * xv * s[k];
        }
        float part = lin + 0.5f * (quad - p2);
        part += __shfl_down(part, 4, 8);
        part += __shfl_down(part, 2, 8);
        part += __shfl_down(part, 1, 8);
        if (c == 0) {
            const int b   = b0 + m;
            const int uid = uid_batch[b];
            const int iid = iid_batch[b];
            out[b] = part + bias_u[uid] + bias_i[iid] + 3.0f;
        }
    }
}

extern "C" void kernel_launch(void* const* d_in, const int* in_sizes, int n_in,
                              void* d_out, int out_size, void* d_ws, size_t ws_size,
                              hipStream_t stream) {
    const int*   uid_batch  = (const int*)d_in[0];
    const int*   iid_batch  = (const int*)d_in[1];
    const int*   n_users_p  = (const int*)d_in[2];
    const float* user_table = (const float*)d_in[3];
    const float* item_table = (const float*)d_in[4];
    const float* word_table = (const float*)d_in[5];
    const float* w_lin      = (const float*)d_in[6];
    const float* V          = (const float*)d_in[7];
    const float* bias_u     = (const float*)d_in[8];
    const float* bias_i     = (const float*)d_in[9];
    float* out = (float*)d_out;

    __bf16* Gb = (__bf16*)d_ws;
    float*  s  = (float*)((char*)d_ws + (size_t)FMD * FMD * sizeof(__bf16));

    // Precompute Gram matrix + column-square sums (tiny: 16.7M MAC)
    fm_prep_kernel<<<FMD, FMD, 0, stream>>>(V, Gb, s);

    const int batch = in_sizes[0];           // 16384
    fm_main_kernel<<<batch / ROWS, 128, 0, stream>>>(
        uid_batch, iid_batch, n_users_p,
        user_table, item_table, word_table,
        w_lin, bias_u, bias_i, Gb, s, out);
}
